// MambaFuse_39848706573418
// MI455X (gfx1250) — matrix operand order repaired and checked by reading, not compiled, not run on hardware
//
#include <hip/hip_runtime.h>
#include <hip/hip_bf16.h>

typedef __bf16 bf16_t;
typedef __attribute__((ext_vector_type(16))) __bf16 v16bf;
typedef __attribute__((ext_vector_type(8)))  __bf16 v8bf;
typedef __attribute__((ext_vector_type(8)))  float  v8f;

// ---------------------------------------------------------------------------
// helpers
// ---------------------------------------------------------------------------
__device__ __forceinline__ float silu_f(float x) {
    return x * (1.0f / (1.0f + __expf(-x)));
}

// A fragment: 16x32 bf16, row-major (K fast), lda elements per row.
// ISA layout: lanes 0-15 -> M=lane, K = {k0..k0+7, k0+16..k0+23}
//             lanes 16-31 -> M=lane-16, K = {k0+8..k0+15, k0+24..k0+31}
__device__ __forceinline__ v16bf load_a_frag(const bf16_t* __restrict__ A, int lda,
                                             int m0, int k0, int lane) {
    int r  = lane & 15;
    int hi = lane >> 4;
    const bf16_t* base = A + (size_t)(m0 + r) * lda + k0 + hi * 8;
    v8bf lo = *reinterpret_cast<const v8bf*>(base);
    v8bf hh = *reinterpret_cast<const v8bf*>(base + 16);
    return __builtin_shufflevector(lo, hh, 0,1,2,3,4,5,6,7,8,9,10,11,12,13,14,15);
}

// B fragment: 32x16 bf16 from pre-transposed weights Wt[N][K] (K fast).
// lanes 0-15 -> N=lane, K = {k0..k0+7, k0+16..k0+23} ; lanes 16-31 -> N=lane-16, K = {k0+8..k0+15, k0+24..k0+31}
__device__ __forceinline__ v16bf load_b_frag(const bf16_t* __restrict__ Wt, int ldb,
                                             int n0, int k0, int lane) {
    int n    = lane & 15;
    int koff = (lane >> 4) * 8;
    const bf16_t* base = Wt + (size_t)(n0 + n) * ldb + k0 + koff;
    v8bf lo = *reinterpret_cast<const v8bf*>(base);
    v8bf hh = *reinterpret_cast<const v8bf*>(base + 16);
    return __builtin_shufflevector(lo, hh, 0,1,2,3,4,5,6,7,8,9,10,11,12,13,14,15);
}

#define WMMA_BF16(a, b, c) \
    __builtin_amdgcn_wmma_f32_16x16x32_bf16(false, (a), false, (b), (short)0, (c), false, false)

// ---------------------------------------------------------------------------
// 1) weight transpose + fp32 -> bf16:   Wt[d][c] = W[c][d]
// ---------------------------------------------------------------------------
__global__ void transpose_w_bf16(const float* __restrict__ W, bf16_t* __restrict__ Wt, int C) {
    int idx = blockIdx.x * 256 + threadIdx.x;
    if (idx < C * C) {
        int d = idx / C, c = idx % C;
        Wt[idx] = (bf16_t)W[(size_t)c * C + d];
    }
}

// ---------------------------------------------------------------------------
// 2) LayerNorm + pack to bf16, pixel-major [t][b*P+p][C]; raw fp32 copy of cur
//    x layout: (B, C, T, H, W);  block handles 32 consecutive pixels of (b,t)
// ---------------------------------------------------------------------------
template <int C>
__global__ void ln_pack(const float* __restrict__ x,
                        const float* __restrict__ gc, const float* __restrict__ bc,
                        const float* __restrict__ gr, const float* __restrict__ br,
                        bf16_t* __restrict__ An, float* __restrict__ curraw,
                        int T, int P, int Ptot) {
    __shared__ float tile[32 * (C + 1)];
    __shared__ float mean_s[32], inv_s[32];

    const int t   = blockIdx.y;
    const int tpi = P / 32;
    const int b   = blockIdx.x / tpi;
    const int p0  = (blockIdx.x % tpi) * 32;
    const int tid = threadIdx.x;

    const size_t strideC = (size_t)T * P;
    const float* xb = x + (size_t)b * C * strideC + (size_t)t * P + p0;

    // coalesced load (32 contiguous pixels per channel row), LDS transpose
    for (int i = tid; i < 32 * C; i += 256) {
        int c = i >> 5, p = i & 31;
        tile[p * (C + 1) + c] = xb[(size_t)c * strideC + p];
    }
    __syncthreads();

    if (tid < 32) {
        const float* row = &tile[tid * (C + 1)];
        float s = 0.0f, s2 = 0.0f;
#pragma unroll 8
        for (int c = 0; c < C; ++c) { float v = row[c]; s += v; s2 += v * v; }
        float m   = s / (float)C;
        float var = s2 / (float)C - m * m;
        mean_s[tid] = m;
        inv_s[tid]  = rsqrtf(var + 1e-5f);
    }
    __syncthreads();

    const bool  isCur = (t == T - 1);
    const float* g  = isCur ? gc : gr;
    const float* bb = isCur ? bc : br;

    const size_t rowBase = ((size_t)t * Ptot + (size_t)b * P + p0) * C;
    for (int i = tid; i < 32 * C; i += 256) {
        int p = i / C, c = i % C;
        float v    = tile[p * (C + 1) + c];
        float norm = (v - mean_s[p]) * inv_s[p] * g[c] + bb[c];
        An[rowBase + (size_t)p * C + c] = (bf16_t)norm;
        if (isCur)
            curraw[((size_t)b * P + p0 + p) * C + c] = v;
    }
}

// ---------------------------------------------------------------------------
// 3) cur_proj = LN(cur) @ Wcur   (fp32 out)
//    wave tile: 16(M) x 64(N); block = 8 waves = 128 pixels x 64 channels
// ---------------------------------------------------------------------------
template <int C>
__global__ void __launch_bounds__(256)
gemm_cur(const bf16_t* __restrict__ A, const bf16_t* __restrict__ Bt,
         float* __restrict__ out) {
    const int lane = threadIdx.x & 31;
    const int wave = threadIdx.x >> 5;
    const int m0 = (blockIdx.x * 8 + wave) * 16;
    const int n0 = blockIdx.y * 64;

    v8f acc[4] = {};
#pragma unroll
    for (int k0 = 0; k0 < C; k0 += 32) {
        v16bf a = load_a_frag(A, C, m0, k0, lane);
#pragma unroll
        for (int nt = 0; nt < 4; ++nt) {
            v16bf b = load_b_frag(Bt, C, n0 + nt * 16, k0, lane);
            acc[nt] = WMMA_BF16(a, b, acc[nt]);
        }
    }
    const int mbase = m0 + ((lane >> 4) << 3);
#pragma unroll
    for (int nt = 0; nt < 4; ++nt) {
        const int n = n0 + nt * 16 + (lane & 15);
#pragma unroll
        for (int r = 0; r < 8; ++r)
            out[(size_t)(mbase + r) * C + n] = acc[nt][r];
    }
}

// ---------------------------------------------------------------------------
// 4) h[t] = cur + silu(LN(ref_t) @ Wref + cur_proj + bias), all 4 t fused:
//    wave tile 16(M) x 64(N) x 4(t); B fragments loaded once per k-step and
//    reused by all four time slices (1.0 b128 load per WMMA).
// ---------------------------------------------------------------------------
template <int C>
__global__ void __launch_bounds__(256)
gemm_ref_h4(const bf16_t* __restrict__ An, const bf16_t* __restrict__ Bt,
            const float* __restrict__ curproj, const float* __restrict__ curraw,
            const float* __restrict__ bias, bf16_t* __restrict__ h, int Ptot) {
    const int lane = threadIdx.x & 31;
    const int wave = threadIdx.x >> 5;
    const int m0 = (blockIdx.x * 8 + wave) * 16;
    const int n0 = blockIdx.y * 64;
    const size_t slice = (size_t)Ptot * C;

    v8f acc[4][4] = {};   // [t][nt]
#pragma unroll
    for (int k0 = 0; k0 < C; k0 += 32) {
        v16bf b[4];
#pragma unroll
        for (int nt = 0; nt < 4; ++nt)
            b[nt] = load_b_frag(Bt, C, n0 + nt * 16, k0, lane);
#pragma unroll
        for (int t = 0; t < 4; ++t) {
            v16bf a = load_a_frag(An + t * slice, C, m0, k0, lane);
#pragma unroll
            for (int nt = 0; nt < 4; ++nt)
                acc[t][nt] = WMMA_BF16(a, b[nt], acc[t][nt]);
        }
    }

    const int mbase = m0 + ((lane >> 4) << 3);
#pragma unroll
    for (int nt = 0; nt < 4; ++nt) {
        const int   n  = n0 + nt * 16 + (lane & 15);
        const float bi = bias[n];
#pragma unroll
        for (int r = 0; r < 8; ++r) {
            const size_t idx = (size_t)(mbase + r) * C + n;
            const float  cp  = curproj[idx] + bi;
            const float  cr  = curraw[idx];
#pragma unroll
            for (int t = 0; t < 4; ++t) {
                float s = acc[t][nt][r] + cp;
                h[t * slice + idx] = (bf16_t)(cr + silu_f(s));
            }
        }
    }
}

// ---------------------------------------------------------------------------
// 5) y_t = silu((h_t @ convw) * bns + bnb); out = max_t(y) + mean_t(y),
//    all 4 t fused in registers; output layout (B, C, H, W)
// ---------------------------------------------------------------------------
template <int C>
__global__ void __launch_bounds__(256)
gemm_conv_pool4(const bf16_t* __restrict__ h, const bf16_t* __restrict__ Bt,
                const float* __restrict__ bns, const float* __restrict__ bnb,
                float* __restrict__ out, int Ptot, int P) {
    const int lane = threadIdx.x & 31;
    const int wave = threadIdx.x >> 5;
    const int m0 = (blockIdx.x * 8 + wave) * 16;
    const int n0 = blockIdx.y * 64;
    const size_t slice = (size_t)Ptot * C;

    v8f acc[4][4] = {};   // [t][nt]
#pragma unroll
    for (int k0 = 0; k0 < C; k0 += 32) {
        v16bf b[4];
#pragma unroll
        for (int nt = 0; nt < 4; ++nt)
            b[nt] = load_b_frag(Bt, C, n0 + nt * 16, k0, lane);
#pragma unroll
        for (int t = 0; t < 4; ++t) {
            v16bf a = load_a_frag(h + t * slice, C, m0, k0, lane);
#pragma unroll
            for (int nt = 0; nt < 4; ++nt)
                acc[t][nt] = WMMA_BF16(a, b[nt], acc[t][nt]);
        }
    }

    const int mbase = m0 + ((lane >> 4) << 3);
#pragma unroll
    for (int nt = 0; nt < 4; ++nt) {
        const int   n  = n0 + nt * 16 + (lane & 15);
        const float sc = bns[n];
        const float sb = bnb[n];
#pragma unroll
        for (int r = 0; r < 8; ++r) {
            float vmax = -3.4e38f, vsum = 0.0f;
#pragma unroll
            for (int t = 0; t < 4; ++t) {
                float y = silu_f(acc[t][nt][r] * sc + sb);
                vmax = fmaxf(vmax, y);
                vsum += y;
            }
            int pix = mbase + r;
            int b   = pix / P;
            int pp  = pix - b * P;
            out[((size_t)b * C + n) * P + pp] = vmax + vsum * 0.25f;
        }
    }
}

// ---------------------------------------------------------------------------
// host-side orchestration
// ---------------------------------------------------------------------------
namespace {

template <int C>
void run_branch(const float* x, const float* gc, const float* bc, const float* gr,
                const float* br, const float* Wcur, const float* Wref, const float* bias,
                const float* convw, const float* bns, const float* bnb, float* out,
                int P, char* ws, size_t& off, hipStream_t stream) {
    const int Bsz  = 8;
    const int T    = 5;
    const int Ptot = Bsz * P;

    auto take = [&](size_t bytes) -> void* {
        off = (off + 255) & ~(size_t)255;
        void* p = ws + off;
        off += bytes;
        return p;
    };

    bf16_t* An     = (bf16_t*)take((size_t)T * Ptot * C * sizeof(bf16_t));
    float*  craw   = (float*) take((size_t)Ptot * C * sizeof(float));
    float*  cproj  = (float*) take((size_t)Ptot * C * sizeof(float));
    bf16_t* hbuf   = (bf16_t*)take((size_t)4 * Ptot * C * sizeof(bf16_t));
    bf16_t* WcurT  = (bf16_t*)take((size_t)C * C * sizeof(bf16_t));
    bf16_t* WrefT  = (bf16_t*)take((size_t)C * C * sizeof(bf16_t));
    bf16_t* convT  = (bf16_t*)take((size_t)C * C * sizeof(bf16_t));

    const int wblocks = (C * C + 255) / 256;
    transpose_w_bf16<<<wblocks, 256, 0, stream>>>(Wcur, WcurT, C);
    transpose_w_bf16<<<wblocks, 256, 0, stream>>>(Wref, WrefT, C);
    transpose_w_bf16<<<wblocks, 256, 0, stream>>>(convw, convT, C);

    ln_pack<C><<<dim3(Ptot / 32, T), 256, 0, stream>>>(x, gc, bc, gr, br, An, craw,
                                                       T, P, Ptot);

    const dim3 ggrid(Ptot / 128, C / 64);
    gemm_cur<C><<<ggrid, 256, 0, stream>>>(An + (size_t)4 * Ptot * C, WcurT, cproj);

    gemm_ref_h4<C><<<ggrid, 256, 0, stream>>>(An, WrefT, cproj, craw, bias, hbuf, Ptot);

    gemm_conv_pool4<C><<<ggrid, 256, 0, stream>>>(hbuf, convT, bns, bnb, out, Ptot, P);
}

} // namespace

extern "C" void kernel_launch(void* const* d_in, const int* in_sizes, int n_in,
                              void* d_out, int out_size, void* d_ws, size_t ws_size,
                              hipStream_t stream) {
    (void)in_sizes; (void)n_in; (void)out_size; (void)ws_size;

    const float* x2 = (const float*)d_in[0];
    const float* x1 = (const float*)d_in[1];
    // d_in[2] (x0) is unused by the reference.

    const float* g2c = (const float*)d_in[3];
    const float* b2c = (const float*)d_in[4];
    const float* g2r = (const float*)d_in[5];
    const float* b2r = (const float*)d_in[6];
    const float* W2cur  = (const float*)d_in[7];
    const float* W2ref  = (const float*)d_in[8];
    const float* bias2  = (const float*)d_in[9];
    const float* convw2 = (const float*)d_in[10];
    const float* bns2   = (const float*)d_in[11];
    const float* bnb2   = (const float*)d_in[12];

    const float* g1c = (const float*)d_in[13];
    const float* b1c = (const float*)d_in[14];
    const float* g1r = (const float*)d_in[15];
    const float* b1r = (const float*)d_in[16];
    const float* W1cur  = (const float*)d_in[17];
    const float* W1ref  = (const float*)d_in[18];
    const float* bias1  = (const float*)d_in[19];
    const float* convw1 = (const float*)d_in[20];
    const float* bns1   = (const float*)d_in[21];
    const float* bnb1   = (const float*)d_in[22];

    float* out2 = (float*)d_out;                          // (8, 128, 64, 64)
    float* out1 = out2 + (size_t)8 * 128 * 64 * 64;       // (8, 256, 32, 32)

    char*  ws  = (char*)d_ws;
    size_t off = 0;

    run_branch<128>(x2, g2c, b2c, g2r, b2r, W2cur, W2ref, bias2, convw2, bns2, bnb2,
                    out2, 64 * 64, ws, off, stream);
    run_branch<256>(x1, g1c, b1c, g1r, b1r, W1cur, W1ref, bias1, convw1, bns1, bnb1,
                    out1, 32 * 32, ws, off, stream);
}
